// ActiNetCRFModel_41480794144883
// MI455X (gfx1250) — matrix operand order repaired and checked by reading, not compiled, not run on hardware
//
#include <hip/hip_runtime.h>
#include <hip/hip_bf16.h>
#include <cstdint>

// ---------------------------------------------------------------------------
// Problem constants (reference: B=128, T=2048, F=2, H=64, C=2)
// ---------------------------------------------------------------------------
#define BB 128
#define TT 2048
#define HH 64
#define NROW (BB * TT)       // 262144 rows of activations
#define EPS 1e-5f

typedef __attribute__((ext_vector_type(16))) _Float16 v16h;
typedef __attribute__((ext_vector_type(8)))  _Float16 v8h;
typedef __attribute__((ext_vector_type(8)))  float    v8f;

#if defined(__gfx1250__) && __has_builtin(__builtin_amdgcn_tensor_load_to_lds)
#define USE_TDM 1
#else
#define USE_TDM 0
#endif

// ---------------------------------------------------------------------------
// Fast transcendental helpers (single TRANS32 ops on CDNA5, branch-free).
// Used on the latency-critical LSTM/CRF serial chains.
// ---------------------------------------------------------------------------
__device__ __forceinline__ float fexp(float x) {
  return __builtin_amdgcn_exp2f(x * 1.4426950408889634f);
}
__device__ __forceinline__ float flog(float x) {
  return __builtin_amdgcn_logf(x) * 0.69314718055994531f;
}
__device__ __forceinline__ float frcp(float x) {
  return __builtin_amdgcn_rcpf(x);
}
__device__ __forceinline__ float fsigm(float x) {
  return frcp(1.0f + fexp(-x));
}
__device__ __forceinline__ float ftanh(float x) {
  float e = fexp(-2.0f * x);
  return (1.0f - e) * frcp(1.0f + e);
}
__device__ __forceinline__ float gelu_exact(float v) {
  return 0.5f * v * (1.0f + erff(v * 0.70710678118654752f));
}

// ---------------------------------------------------------------------------
// WMMA helpers (CDNA5 16x16x32 f16 -> f32)
// ---------------------------------------------------------------------------
__device__ __forceinline__ v8f wmma16(v16h a, v16h b, v8f c) {
#if defined(__gfx1250__)
  return __builtin_amdgcn_wmma_f32_16x16x32_f16(
      /*neg_a=*/false, a, /*neg_b=*/false, b,
      /*c_mod=*/(short)0, c, /*reuse_a=*/false, /*reuse_b=*/false);
#else
  (void)a; (void)b; return c;
#endif
}

// A fragment: 16x32 f16 tile, row-major in LDS with element stride `stride`.
// lanes 0-15:  row=lane,    K = k0+{0..15}
// lanes 16-31: row=lane-16, K = k0+16+{0..15}
__device__ __forceinline__ v16h load_a_frag(const _Float16* base, int lane,
                                            int k0, int stride) {
  int row  = lane & 15;
  int koff = k0 + ((lane >> 4) << 4);
  const _Float16* p = base + row * stride + koff;
  v8h lo = *(const v8h*)(p);
  v8h hi = *(const v8h*)(p + 8);
  v16h a;
#pragma unroll
  for (int j = 0; j < 8; ++j) { a[j] = lo[j]; a[8 + j] = hi[j]; }
  return a;
}

// B fragments pre-swizzled in LDS in fragment order:
// dst[((nt*2 + kh)*32 + lane)*16 + j]  holds  W^T[k, n] = W[n*KDIM + k] with
// n = nt*16 + (lane&15), k = kh*32 + ((lane>>4)<<4) + j.
// A fragment load is then two contiguous ds_load_b128 per lane.
template <int KDIM>
__device__ __forceinline__ void stage_bfrags(_Float16* dst,
                                             const float* __restrict__ W,
                                             int nelem, int tid, int nthr) {
  for (int i = tid; i < nelem; i += nthr) {
    int j    = i & 15;
    int lane = (i >> 4) & 31;
    int kh   = (i >> 9) & 1;
    int nt   = i >> 10;
    int n = nt * 16 + (lane & 15);
    int k = kh * 32 + ((lane >> 4) << 4) + j;
    dst[i] = (_Float16)W[n * KDIM + k];
  }
}

__device__ __forceinline__ v16h load_b_frag_sw(const _Float16* base, int lane,
                                               int fragid) {
  const _Float16* p = base + ((size_t)(fragid * 32 + lane)) * 16;
  v8h lo = *(const v8h*)(p);
  v8h hi = *(const v8h*)(p + 8);
  v16h b;
#pragma unroll
  for (int j = 0; j < 8; ++j) { b[j] = lo[j]; b[8 + j] = hi[j]; }
  return b;
}

// ---------------------------------------------------------------------------
// Kernel 1: fused conv stack: 4x (1x1 conv -> exact GELU -> eval BatchNorm).
// Output written transposed to [T, B, H] in f16 (LSTM-ready).
// grid = NROW/128 blocks x 256 threads (8 waves).
// ---------------------------------------------------------------------------
__global__ __launch_bounds__(256) void conv_stack_kernel(
    const float* __restrict__ x, const float* __restrict__ cw1,
    const float* __restrict__ cb1, const float* __restrict__ cwr,
    const float* __restrict__ cbr, const float* __restrict__ bn_g,
    const float* __restrict__ bn_b, const float* __restrict__ bn_m,
    const float* __restrict__ bn_v, _Float16* __restrict__ yout) {
  __shared__ __align__(16) _Float16 actA[128 * 72];
  __shared__ __align__(16) _Float16 actB[128 * 72];
  __shared__ __align__(16) _Float16 wT[4 * 2 * 32 * 16];  // 8 frags (4 nt, 2 kh)
  __shared__ float sscale[4 * 64];
  __shared__ float sshift[4 * 64];

  const int tid  = threadIdx.x;
  const int wave = tid >> 5, lane = tid & 31;
  const int bt0  = blockIdx.x * 128;

  // Fold BN (eval) into scale/shift per layer.
  for (int i = tid; i < 4 * 64; i += 256) {
    float inv  = rsqrtf(bn_v[i] + EPS);
    float sc   = bn_g[i] * inv;
    sscale[i]  = sc;
    sshift[i]  = bn_b[i] - bn_m[i] * sc;
  }
  __syncthreads();

  // Layer 1: F=2 -> H=64 (trivial, scalar), then GELU + BN.
  for (int i = tid; i < 128 * 64; i += 256) {
    int r = i >> 6, h = i & 63;
    int bt = bt0 + r;
    float v = x[bt * 2 + 0] * cw1[h * 2 + 0] + x[bt * 2 + 1] * cw1[h * 2 + 1] +
              cb1[h];
    v = gelu_exact(v);
    v = v * sscale[h] + sshift[h];
    actA[r * 72 + h] = (_Float16)v;
  }
  __syncthreads();

  _Float16* cur = actA;
  _Float16* nxt = actB;
  for (int l = 0; l < 3; ++l) {
    stage_bfrags<64>(wT, cwr + l * 64 * 64, 4 * 1024, tid, 256);
    __syncthreads();

    const int mt = wave;  // 8 waves -> 8 M-tiles of 16 rows
    v16h a0 = load_a_frag(cur + mt * 16 * 72, lane, 0, 72);
    v16h a1 = load_a_frag(cur + mt * 16 * 72, lane, 32, 72);
    const int col16 = lane & 15;
    const int rbase = mt * 16 + ((lane >> 4) << 3);

    // Batched: load all B-fragments, issue 4 independent WMMA chains
    // back-to-back (hazard window covered by chain interleave), then one
    // epilogue pass.  No v_nop stalls between matrix ops.
    v16h b0[4], b1[4];
    v8f acc[4];
#pragma unroll
    for (int nt = 0; nt < 4; ++nt) {
      b0[nt] = load_b_frag_sw(wT, lane, nt * 2 + 0);
      b1[nt] = load_b_frag_sw(wT, lane, nt * 2 + 1);
      float bias = cbr[l * 64 + nt * 16 + col16];
      acc[nt] = (v8f){bias, bias, bias, bias, bias, bias, bias, bias};
    }
#pragma unroll
    for (int nt = 0; nt < 4; ++nt) acc[nt] = wmma16(a0, b0[nt], acc[nt]);
#pragma unroll
    for (int nt = 0; nt < 4; ++nt) acc[nt] = wmma16(a1, b1[nt], acc[nt]);

    __syncthreads();  // everyone done reading `cur`/wT before rewrites
#pragma unroll
    for (int nt = 0; nt < 4; ++nt) {
      int col = nt * 16 + col16;
      float sc = sscale[(l + 1) * 64 + col];
      float sh = sshift[(l + 1) * 64 + col];
#pragma unroll
      for (int r = 0; r < 8; ++r) {
        float v = gelu_exact(acc[nt][r]) * sc + sh;
        nxt[(rbase + r) * 72 + col] = (_Float16)v;
      }
    }
    __syncthreads();
    _Float16* t = cur; cur = nxt; nxt = t;
  }

  // Store transposed: row bt = b*T + t  ->  yout[(t*B + b)*64 + h]
  for (int i = tid; i < 128 * 64; i += 256) {
    int r = i >> 6, h = i & 63;
    int bt = bt0 + r;
    int b = bt >> 11, t = bt & (TT - 1);
    yout[((size_t)t * BB + b) * 64 + h] = cur[r * 72 + h];
  }
}

// ---------------------------------------------------------------------------
// Kernel 2: gate-input GEMM for both LSTM directions.
// Z is written in C-FRAGMENT order:
//   Z[(((tile16*16) + nt)*32 + lane)*8 + r], tile16 = global_row/16.
// A 16-row slab (one tile16) is a contiguous 16KB block -> TDM friendly, and
// the recurrence reads each lane's 8 accumulator values as one v8f.
// grid = NROW/128 blocks x 256 threads.
// ---------------------------------------------------------------------------
__global__ __launch_bounds__(256) void gate_input_gemm(
    const _Float16* __restrict__ y, const float* __restrict__ wih_f,
    const float* __restrict__ bih_f, const float* __restrict__ bhh_f,
    const float* __restrict__ wih_b, const float* __restrict__ bih_b,
    const float* __restrict__ bhh_b, float* __restrict__ Zf,
    float* __restrict__ Zb) {
  __shared__ __align__(16) _Float16 act[128 * 72];
  __shared__ __align__(16) _Float16 wfragF[16 * 2 * 32 * 16];  // 32KB
  __shared__ __align__(16) _Float16 wfragB[16 * 2 * 32 * 16];  // 32KB

  const int tid  = threadIdx.x;
  const int wave = tid >> 5, lane = tid & 31;
  const int row0 = blockIdx.x * 128;

  for (int i = tid; i < 128 * 64; i += 256) {
    int r = i >> 6, h = i & 63;
    act[r * 72 + h] = y[((size_t)row0 + r) * 64 + h];
  }
  stage_bfrags<64>(wfragF, wih_f, 16 * 1024, tid, 256);
  stage_bfrags<64>(wfragB, wih_b, 16 * 1024, tid, 256);
  __syncthreads();

  const int mt = wave;
  v16h a0 = load_a_frag(act + mt * 16 * 72, lane, 0, 72);
  v16h a1 = load_a_frag(act + mt * 16 * 72, lane, 32, 72);
  const int col16 = lane & 15;
  const size_t tile16 = (size_t)(row0 / 16 + mt);

  for (int d = 0; d < 2; ++d) {
    const _Float16* wt = d ? wfragB : wfragF;
    const float* bi = d ? bih_b : bih_f;
    const float* bh = d ? bhh_b : bhh_f;
    float* Z = d ? Zb : Zf;
#pragma unroll 1
    for (int ntq = 0; ntq < 16; ntq += 4) {
      v16h b0[4], b1[4];
      v8f acc[4];
#pragma unroll
      for (int q = 0; q < 4; ++q) {
        int nt = ntq + q;
        b0[q] = load_b_frag_sw(wt, lane, nt * 2 + 0);
        b1[q] = load_b_frag_sw(wt, lane, nt * 2 + 1);
        int col = nt * 16 + col16;
        float bias = bi[col] + bh[col];
        acc[q] = (v8f){bias, bias, bias, bias, bias, bias, bias, bias};
      }
#pragma unroll
      for (int q = 0; q < 4; ++q) acc[q] = wmma16(a0, b0[q], acc[q]);
#pragma unroll
      for (int q = 0; q < 4; ++q) acc[q] = wmma16(a1, b1[q], acc[q]);
#pragma unroll
      for (int q = 0; q < 4; ++q) {
        float* zp = Z + (((tile16 * 16 + (ntq + q)) * 32 + lane) * 8);
        *(v8f*)zp = acc[q];  // two global_store_b128
      }
    }
  }
}

// ---------------------------------------------------------------------------
// TDM prefetch of a contiguous 16x256 f32 slab (16KB) into LDS.
// ---------------------------------------------------------------------------
#if USE_TDM
typedef unsigned int u32x4_t __attribute__((ext_vector_type(4)));
typedef int i32x4_t __attribute__((ext_vector_type(4)));
typedef int i32x8_t __attribute__((ext_vector_type(8)));

__device__ __forceinline__ void tdm_load_16x256(const float* src, float* dst) {
  unsigned long long ga = (unsigned long long)(uintptr_t)src;
  unsigned int la = (unsigned int)(uintptr_t)dst;  // low 32b = LDS offset
  u32x4_t g0;
  g0.x = 1u;                                        // count=1 (valid D#)
  g0.y = la;                                        // lds_addr
  g0.z = (unsigned int)ga;                          // global_addr[31:0]
  g0.w = (unsigned int)(ga >> 32) | (2u << 30);     // global hi | type=2
  i32x8_t g1;
  g1[0] = (2 << 16);                       // data_size = 4B; no multicast/pad
  g1[1] = (int)((4096u & 0xFFFFu) << 16);  // tensor_dim0 = 4096 (lo16)
  g1[2] = (int)(1u << 16);                 // tensor_dim1 = 1
  g1[3] = (int)(4096u << 16);              // tile_dim0 = 4096 elems (1-D tile)
  g1[4] = 0;                               // tile_dim1/2 unused
  g1[5] = 4096;                            // tensor_dim0_stride
  g1[6] = 0;
  g1[7] = 0;
  i32x4_t gz = {0, 0, 0, 0};
#if defined(__clang_major__) && (__clang_major__ >= 23)
  i32x8_t gz8 = {0, 0, 0, 0, 0, 0, 0, 0};
  __builtin_amdgcn_tensor_load_to_lds(g0, g1, gz, gz, gz8, 0);
#else
  __builtin_amdgcn_tensor_load_to_lds(g0, g1, gz, gz, 0);
#endif
}
#endif

// ---------------------------------------------------------------------------
// Kernel 3: LSTM recurrence, both directions.
// grid = 16 blocks (dir * 8 batch-slices of 16), block = 128 threads (4 waves).
// Wave j owns hidden columns [16j, 16j+16): gate tiles {j, j+4, j+8, j+12}.
// Whh^T B-fragments live in registers for the whole T loop; h state in LDS;
// c state in VGPRs. Next step's Z slab is TDM-prefetched into LDS.
// ---------------------------------------------------------------------------
__global__ __launch_bounds__(128) void lstm_rec_kernel(
    const float* __restrict__ Zf, const float* __restrict__ Zb,
    const float* __restrict__ whh_f, const float* __restrict__ whh_b,
    _Float16* __restrict__ hf, _Float16* __restrict__ hb) {
  const int dir   = blockIdx.x >> 3;
  const int slice = blockIdx.x & 7;
  const int b0    = slice * 16;
  const float* Z    = dir ? Zb : Zf;
  const float* whh  = dir ? whh_b : whh_f;
  _Float16*    hout = dir ? hb : hf;

  __shared__ __align__(16) _Float16 hbuf[16 * 72];
  __shared__ __align__(16) float zbuf[2][16 * 256];

  const int tid = threadIdx.x, wave = tid >> 5, lane = tid & 31;
  const int col16 = lane & 15;
  const int mhi   = (lane >> 4) << 3;

  // Load Whh^T B-fragments into registers: B[k][n] = whh[n*64 + k].
  v16h Bf[4][2];
#pragma unroll
  for (int g = 0; g < 4; ++g) {
    int nt = wave + 4 * g;
    int n  = nt * 16 + col16;
#pragma unroll
    for (int kh = 0; kh < 2; ++kh) {
      int kb = kh * 32 + ((lane >> 4) << 4);
      v16h b;
#pragma unroll
      for (int j = 0; j < 16; ++j) b[j] = (_Float16)whh[n * 64 + kb + j];
      Bf[g][kh] = b;
    }
  }

  for (int i = tid; i < 16 * 72; i += 128) hbuf[i] = (_Float16)0.0f;
  float cst[8];
#pragma unroll
  for (int r = 0; r < 8; ++r) cst[r] = 0.0f;

  // Prime Z slab for step 0 (tile16 = t*8 + b0/16, contiguous 16KB).
  {
    int t0 = dir ? (TT - 1) : 0;
    const float* src = Z + ((size_t)t0 * 8 + (b0 >> 4)) * 4096;
#if USE_TDM
    if (wave == 0 && lane == 0) tdm_load_16x256(src, &zbuf[0][0]);
#else
    for (int i = tid; i < 16 * 256; i += 128) zbuf[0][i] = src[i];
#endif
  }

#pragma unroll 1
  for (int s = 0; s < TT; ++s) {
    const int t   = dir ? (TT - 1 - s) : s;
    const int cur = s & 1;
    const int nx  = cur ^ 1;

#if USE_TDM
    if (wave == 0) __builtin_amdgcn_s_wait_tensorcnt(0);
#endif
    __syncthreads();  // zbuf[cur] ready; hbuf from step s-1 visible

    // Kick prefetch of step s+1 into the other buffer.
    if (s + 1 < TT) {
      int t2 = dir ? (TT - 2 - s) : (s + 1);
      const float* src = Z + ((size_t)t2 * 8 + (b0 >> 4)) * 4096;
#if USE_TDM
      if (wave == 0 && lane == 0) tdm_load_16x256(src, &zbuf[nx][0]);
#else
      for (int i = tid; i < 16 * 256; i += 128) zbuf[nx][i] = src[i];
#endif
    }

    // Accumulators start at Z (fragment layout: one v8f per gate tile).
    v8f acc[4];
#pragma unroll
    for (int g = 0; g < 4; ++g) {
      int nt = wave + 4 * g;
      acc[g] = *(const v8f*)&zbuf[cur][(size_t)(nt * 32 + lane) * 8];
    }

    // h @ Whh^T  (A from LDS h state). kh-outer: 4 independent WMMA chains.
    v16h a0 = load_a_frag(hbuf, lane, 0, 72);
    v16h a1 = load_a_frag(hbuf, lane, 32, 72);
    __syncthreads();  // all A-frag reads done before hbuf is rewritten
    acc[0] = wmma16(a0, Bf[0][0], acc[0]);
    acc[1] = wmma16(a0, Bf[1][0], acc[1]);
    acc[2] = wmma16(a0, Bf[2][0], acc[2]);
    acc[3] = wmma16(a0, Bf[3][0], acc[3]);
    acc[0] = wmma16(a1, Bf[0][1], acc[0]);
    acc[1] = wmma16(a1, Bf[1][1], acc[1]);
    acc[2] = wmma16(a1, Bf[2][1], acc[2]);
    acc[3] = wmma16(a1, Bf[3][1], acc[3]);

    // Cell update (PyTorch gate order i,f,g,o), write h to LDS + global.
    const int hcol = 16 * wave + col16;
#pragma unroll
    for (int r = 0; r < 8; ++r) {
      int m = mhi + r;
      float ig = fsigm(acc[0][r]);
      float fg = fsigm(acc[1][r]);
      float gg = ftanh(acc[2][r]);
      float og = fsigm(acc[3][r]);
      float c  = fg * cst[r] + ig * gg;
      cst[r]   = c;
      float h  = og * ftanh(c);
      _Float16 h16 = (_Float16)h;
      hbuf[m * 72 + hcol] = h16;
      hout[((size_t)t * BB + b0 + m) * 64 + hcol] = h16;
    }
  }
}

// ---------------------------------------------------------------------------
// Kernel 4: emissions  em[b,t,c] = [hf;hb] @ fc_w^T + fc_b   (C=2)
// ---------------------------------------------------------------------------
__global__ __launch_bounds__(256) void emission_kernel(
    const _Float16* __restrict__ hf, const _Float16* __restrict__ hb,
    const float* __restrict__ fc_w, const float* __restrict__ fc_b,
    float* __restrict__ em) {
  __shared__ float w[2][128];
  const int tid = threadIdx.x;
  w[tid >> 7][tid & 127] = fc_w[tid & 255];
  __syncthreads();

  size_t idx = (size_t)blockIdx.x * 256 + tid;  // idx = b*T + t
  int b = (int)(idx >> 11), t = (int)(idx & (TT - 1));
  const v8h* pf = (const v8h*)(hf + ((size_t)t * BB + b) * 64);
  const v8h* pb = (const v8h*)(hb + ((size_t)t * BB + b) * 64);
  float e0 = fc_b[0], e1 = fc_b[1];
#pragma unroll
  for (int q = 0; q < 8; ++q) {
    v8h vf = pf[q], vb = pb[q];
#pragma unroll
    for (int j = 0; j < 8; ++j) {
      int k = q * 8 + j;
      float a = (float)vf[j], bv = (float)vb[j];
      e0 += a * w[0][k] + bv * w[0][64 + k];
      e1 += a * w[1][k] + bv * w[1][64 + k];
    }
  }
  em[idx * 2 + 0] = e0;
  em[idx * 2 + 1] = e1;
}

// ---------------------------------------------------------------------------
// Kernel 5: CRF forward algorithm + gold score + NLL sum. 1 block, thread=batch.
// ---------------------------------------------------------------------------
__global__ __launch_bounds__(128) void crf_kernel(
    const float* __restrict__ em, const int* __restrict__ tags,
    const float* __restrict__ cs, const float* __restrict__ ce,
    const float* __restrict__ ct, float* __restrict__ out) {
  __shared__ float red[128];
  const int b = threadIdx.x;
  const float t00 = ct[0], t01 = ct[1], t10 = ct[2], t11 = ct[3];
  const float* e  = em + (size_t)b * TT * 2;
  const int*   tg = tags + (size_t)b * TT;

  float a0 = cs[0] + e[0], a1 = cs[1] + e[1];
  int   prev = tg[0];
  float num  = cs[prev] + e[prev];

#pragma unroll 1
  for (int t = 1; t < TT; ++t) {
    float e0 = e[t * 2 + 0], e1 = e[t * 2 + 1];
    float x0 = a0 + t00, y0 = a1 + t10;
    float m0 = fmaxf(x0, y0);
    float n0 = m0 + flog(fexp(x0 - m0) + fexp(y0 - m0)) + e0;
    float x1 = a0 + t01, y1 = a1 + t11;
    float m1 = fmaxf(x1, y1);
    float n1 = m1 + flog(fexp(x1 - m1) + fexp(y1 - m1)) + e1;
    a0 = n0; a1 = n1;
    int cu = tg[t];
    num += e[t * 2 + cu] + ct[prev * 2 + cu];
    prev = cu;
  }
  num += ce[prev];
  float x = a0 + ce[0], y = a1 + ce[1];
  float m = fmaxf(x, y);
  float lz = m + flog(fexp(x - m) + fexp(y - m));

  red[b] = lz - num;
  __syncthreads();
  for (int s2 = 64; s2 > 0; s2 >>= 1) {
    if (b < s2) red[b] += red[b + s2];
    __syncthreads();
  }
  if (b == 0) out[0] = red[0];
}

// ---------------------------------------------------------------------------
extern "C" void kernel_launch(void* const* d_in, const int* in_sizes, int n_in,
                              void* d_out, int out_size, void* d_ws,
                              size_t ws_size, hipStream_t stream) {
  (void)in_sizes; (void)n_in; (void)out_size; (void)ws_size;
  const float* x     = (const float*)d_in[0];
  const int*   tags  = (const int*)d_in[1];
  const float* cw1   = (const float*)d_in[2];
  const float* cb1   = (const float*)d_in[3];
  const float* cwr   = (const float*)d_in[4];
  const float* cbr   = (const float*)d_in[5];
  const float* bn_g  = (const float*)d_in[6];
  const float* bn_b  = (const float*)d_in[7];
  const float* bn_m  = (const float*)d_in[8];
  const float* bn_v  = (const float*)d_in[9];
  const float* wih_f = (const float*)d_in[10];
  const float* whh_f = (const float*)d_in[11];
  const float* bih_f = (const float*)d_in[12];
  const float* bhh_f = (const float*)d_in[13];
  const float* wih_b = (const float*)d_in[14];
  const float* whh_b = (const float*)d_in[15];
  const float* bih_b = (const float*)d_in[16];
  const float* bhh_b = (const float*)d_in[17];
  const float* fc_w  = (const float*)d_in[18];
  const float* fc_b  = (const float*)d_in[19];
  const float* cs    = (const float*)d_in[20];
  const float* ce    = (const float*)d_in[21];
  const float* ct    = (const float*)d_in[22];

  char* ws = (char*)d_ws;
  size_t off = 0;
  _Float16* y16 = (_Float16*)(ws + off); off += (size_t)NROW * 64 * sizeof(_Float16);
  float*    Zf  = (float*)(ws + off);    off += (size_t)NROW * 256 * sizeof(float);
  float*    Zb  = (float*)(ws + off);    off += (size_t)NROW * 256 * sizeof(float);
  _Float16* hf  = (_Float16*)(ws + off); off += (size_t)NROW * 64 * sizeof(_Float16);
  _Float16* hb  = (_Float16*)(ws + off); off += (size_t)NROW * 64 * sizeof(_Float16);
  float*    em  = (float*)(ws + off);    off += (size_t)NROW * 2 * sizeof(float);

  conv_stack_kernel<<<NROW / 128, 256, 0, stream>>>(
      x, cw1, cb1, cwr, cbr, bn_g, bn_b, bn_m, bn_v, y16);
  gate_input_gemm<<<NROW / 128, 256, 0, stream>>>(
      y16, wih_f, bih_f, bhh_f, wih_b, bih_b, bhh_b, Zf, Zb);
  lstm_rec_kernel<<<16, 128, 0, stream>>>(Zf, Zb, whh_f, whh_b, hf, hb);
  emission_kernel<<<NROW / 256, 256, 0, stream>>>(hf, hb, fc_w, fc_b, em);
  crf_kernel<<<1, 128, 0, stream>>>(em, tags, cs, ce, ct, (float*)d_out);
}
